// SpitalCrossAttention_32968168964364
// MI455X (gfx1250) — matrix operand order repaired and checked by reading, not compiled, not run on hardware
//
#include <hip/hip_runtime.h>
#include <hip/hip_bf16.h>

#define B_  4
#define C_  256
#define HW_ 4096

typedef __attribute__((ext_vector_type(16))) __bf16 v16bf;
typedef __attribute__((ext_vector_type(8)))  float  v8f;
using u16 = unsigned short;

// float -> bf16 (round to nearest even)
static __device__ __forceinline__ u16 f2bf_u(float f) {
  union { float f; unsigned u; } a; a.f = f;
  unsigned u = a.u;
  u += 0x7fffu + ((u >> 16) & 1u);
  return (u16)(u >> 16);
}

union FragB16 { uint4 q[2]; v16bf v; u16 s[16]; };

// A-frag (16x32, M=lane%16): halves {a0..a0+7} U {a0+16..a0+23}, a0 = hi*8
static __device__ __forceinline__ v16bf loadA_bf16(const u16* chunk, int hi) {
  FragB16 f;
  f.q[0] = *(const uint4*)(chunk + hi * 8);
  f.q[1] = *(const uint4*)(chunk + hi * 8 + 16);
  return f.v;
}
// B-frag (32x16, N=lane%16): halves {k0..k0+7} U {k0+16..k0+23}, k0 = hi*8
static __device__ __forceinline__ v16bf loadB_bf16(const u16* chunk, int hi) {
  FragB16 f;
  f.q[0] = *(const uint4*)(chunk + hi * 8);
  f.q[1] = *(const uint4*)(chunk + hi * 8 + 16);
  return f.v;
}
static __device__ __forceinline__ v8f wmma_bf16(v16bf a, v16bf b, v8f c) {
  return __builtin_amdgcn_wmma_f32_16x16x32_bf16(false, a, false, b, (short)0, c,
                                                 false, false);
}
static __device__ __forceinline__ v8f zero8() {
  v8f z;
#pragma unroll
  for (int i = 0; i < 8; i++) z[i] = 0.f;
  return z;
}

// One-shot: convert the four 256x256 fp32 weight matrices to bf16.
__global__ __launch_bounds__(256) void k_cvtw(const float* __restrict__ w0,
                                              const float* __restrict__ w1,
                                              const float* __restrict__ w2,
                                              const float* __restrict__ w3,
                                              u16* __restrict__ out) {
  int e = (blockIdx.x * 256 + threadIdx.x) * 4;  // 16384 threads x 4 elems
  ushort4 o;
  const float4 a = *(const float4*)(w0 + e);
  o = make_ushort4(f2bf_u(a.x), f2bf_u(a.y), f2bf_u(a.z), f2bf_u(a.w));
  *(ushort4*)(out + e) = o;
  const float4 b = *(const float4*)(w1 + e);
  o = make_ushort4(f2bf_u(b.x), f2bf_u(b.y), f2bf_u(b.z), f2bf_u(b.w));
  *(ushort4*)(out + 65536 + e) = o;
  const float4 c = *(const float4*)(w2 + e);
  o = make_ushort4(f2bf_u(c.x), f2bf_u(c.y), f2bf_u(c.z), f2bf_u(c.w));
  *(ushort4*)(out + 131072 + e) = o;
  const float4 d = *(const float4*)(w3 + e);
  o = make_ushort4(f2bf_u(d.x), f2bf_u(d.y), f2bf_u(d.z), f2bf_u(d.w));
  *(ushort4*)(out + 196608 + e) = o;
}

// out = W @ in (+bias), optionally transposed output, bf16 results.
// transpose_out=1: out[b][p][o] = (sum_c in[b][c][p] * w[o][c] + bias[o]) * mul
// transpose_out=0: out[b][o][p] =  sum_c in[b][c][p] * w[o][c] + bias[o]
__global__ __launch_bounds__(128) void k_proj(const float* __restrict__ in,
                                              const u16* __restrict__ wb,
                                              const float* __restrict__ bias,
                                              u16* __restrict__ out,
                                              int transpose_out, float mul) {
  __shared__ alignas(16) u16 xT[64][264];   // in^T tile [p][c], padded
  const int b  = blockIdx.y;
  const int p0 = blockIdx.x * 64;
  const int tid = threadIdx.x;
  const int wv_ = tid >> 5, L = tid & 31, hi = L >> 4, lm = L & 15;

  // stage transposed, float4 global loads
  for (int e = tid; e < 64 * 64; e += 128) {
    int c = e >> 4, p4 = (e & 15) * 4;
    const float4 f = *(const float4*)&in[((size_t)b * C_ + c) * HW_ + p0 + p4];
    xT[p4 + 0][c] = f2bf_u(f.x);
    xT[p4 + 1][c] = f2bf_u(f.y);
    xT[p4 + 2][c] = f2bf_u(f.z);
    xT[p4 + 3][c] = f2bf_u(f.w);
  }
  __syncthreads();

  if (transpose_out) {
    // M = p (wave's 16 rows), N = o (16 tiles), K = c
    const u16* arow = &xT[wv_ * 16 + lm][0];
    v16bf Ax[8];
#pragma unroll
    for (int kc = 0; kc < 8; kc++) Ax[kc] = loadA_bf16(arow + kc * 32, hi);
#pragma unroll
    for (int n = 0; n < 16; n++) {
      const u16* wrow = wb + (size_t)(n * 16 + lm) * C_;
      v16bf Bw[8];
#pragma unroll
      for (int kc = 0; kc < 8; kc++) Bw[kc] = loadB_bf16(wrow + kc * 32, hi);
      v8f acc = zero8();
#pragma unroll
      for (int kc = 0; kc < 8; kc++) acc = wmma_bf16(Ax[kc], Bw[kc], acc);
      float bb = bias[n * 16 + lm];
#pragma unroll
      for (int r = 0; r < 8; r++) {
        float val = (acc[r] + bb) * mul;
        out[((size_t)b * HW_ + p0 + wv_ * 16 + r + hi * 8) * C_ + n * 16 + lm] =
            f2bf_u(val);
      }
    }
  } else {
    // M = o (16 tiles), N = p (wave's 16 cols), K = c
    const u16* brow = &xT[wv_ * 16 + lm][0];
    v16bf Bx[8];
#pragma unroll
    for (int kc = 0; kc < 8; kc++) Bx[kc] = loadB_bf16(brow + kc * 32, hi);
#pragma unroll
    for (int m = 0; m < 16; m++) {
      const u16* wrow = wb + (size_t)(m * 16 + lm) * C_;
      v16bf Aw[8];
#pragma unroll
      for (int kc = 0; kc < 8; kc++) Aw[kc] = loadA_bf16(wrow + kc * 32, hi);
      v8f acc = zero8();
#pragma unroll
      for (int kc = 0; kc < 8; kc++) acc = wmma_bf16(Aw[kc], Bx[kc], acc);
#pragma unroll
      for (int r = 0; r < 8; r++) {
        int o = m * 16 + r + hi * 8;
        out[((size_t)b * C_ + o) * HW_ + p0 + wv_ * 16 + lm] =
            f2bf_u(acc[r] + bias[o]);
      }
    }
  }
}

// Flash attention: qt[b][i][c] (pre-scaled), kt[b][j][c], vv[b][c][j] -> ot[b][i][c]
__global__ __launch_bounds__(128) void k_attn(const u16* __restrict__ qt,
                                              const u16* __restrict__ kt,
                                              const u16* __restrict__ vv,
                                              u16* __restrict__ ot) {
  __shared__ alignas(16) u16 qs[4][16][264];  // per-wave Q rows
  __shared__ alignas(16) u16 ks[32][264];     // K tile [j][c]
  __shared__ alignas(16) u16 vs[256][40];     // V tile [c][j]
  __shared__ alignas(16) u16 ps[4][16][40];   // per-wave P tile [i][j]
  const int b  = blockIdx.y;
  const int i0 = blockIdx.x * 64;
  const int tid = threadIdx.x;
  const int wv_ = tid >> 5, L = tid & 31, hi = L >> 4, lm = L & 15;

  for (int e = tid; e < 64 * 32; e += 128) {        // stage Q (16B chunks)
    int row = e >> 5, c8 = (e & 31) * 8;
    *(uint4*)&qs[row >> 4][row & 15][c8] =
        *(const uint4*)&qt[((size_t)b * HW_ + i0 + row) * C_ + c8];
  }

  v8f O[16];
#pragma unroll
  for (int t = 0; t < 16; t++) O[t] = zero8();
  float m_[8], l_[8];
#pragma unroll
  for (int r = 0; r < 8; r++) { m_[r] = -3.0e38f; l_[r] = 0.f; }
  const float LOG2E = 1.44269504088896f;

  for (int jc = 0; jc < 128; jc++) {
    const int j0 = jc * 32;
    for (int e = tid; e < 512; e += 128) {          // stage K tile
      int jj = e >> 4, c16 = (e & 15) * 16;
      *(uint4*)&ks[jj][c16] =
          *(const uint4*)&kt[((size_t)b * HW_ + j0 + jj) * C_ + c16];
    }
    for (int e = tid; e < 1024; e += 128) {         // stage V tile
      int c = e >> 2, j8 = (e & 3) * 8;
      *(uint4*)&vs[c][j8] =
          *(const uint4*)&vv[((size_t)b * C_ + c) * HW_ + j0 + j8];
    }
    // prefetch next chunk (global_prefetch_b8 path)
    if (jc + 1 < 128) {
      const int jn = j0 + 32;
      if (tid < 32)
        __builtin_prefetch(&kt[((size_t)b * HW_ + jn + tid) * C_], 0, 3);
      __builtin_prefetch(&vv[((size_t)b * C_ + tid * 2) * HW_ + jn], 0, 3);
      __builtin_prefetch(&vv[((size_t)b * C_ + tid * 2 + 1) * HW_ + jn], 0, 3);
    }
    __syncthreads();

    // S = Q' K'^T for two 16-col subtiles
    v8f S0 = zero8(), S1 = zero8();
    const u16* qrow = &qs[wv_][lm][0];
#pragma unroll
    for (int kc = 0; kc < 8; kc += 2) {
      v16bf A0  = loadA_bf16(qrow + kc * 32, hi);
      v16bf A1  = loadA_bf16(qrow + kc * 32 + 32, hi);
      v16bf B00 = loadB_bf16(&ks[lm][kc * 32], hi);
      v16bf B01 = loadB_bf16(&ks[16 + lm][kc * 32], hi);
      v16bf B10 = loadB_bf16(&ks[lm][kc * 32 + 32], hi);
      v16bf B11 = loadB_bf16(&ks[16 + lm][kc * 32 + 32], hi);
      S0 = wmma_bf16(A0, B00, S0);
      S1 = wmma_bf16(A0, B01, S1);
      S0 = wmma_bf16(A1, B10, S0);
      S1 = wmma_bf16(A1, B11, S1);
    }

    // online softmax (rows live inside one 16-lane half-wave)
    float alpha[8];
#pragma unroll
    for (int r = 0; r < 8; r++) {
      float mx = fmaxf(S0[r], S1[r]);
      mx = fmaxf(mx, __shfl_xor(mx, 1, 16));
      mx = fmaxf(mx, __shfl_xor(mx, 2, 16));
      mx = fmaxf(mx, __shfl_xor(mx, 4, 16));
      mx = fmaxf(mx, __shfl_xor(mx, 8, 16));
      float mN = fmaxf(m_[r], mx);
      float a  = exp2f((m_[r] - mN) * LOG2E);
      float p0 = exp2f((S0[r] - mN) * LOG2E);
      float p1 = exp2f((S1[r] - mN) * LOG2E);
      float rs = p0 + p1;
      rs += __shfl_xor(rs, 1, 16);
      rs += __shfl_xor(rs, 2, 16);
      rs += __shfl_xor(rs, 4, 16);
      rs += __shfl_xor(rs, 8, 16);
      m_[r] = mN;
      l_[r] = l_[r] * a + rs;
      alpha[r] = a;
      ps[wv_][r + hi * 8][lm]      = f2bf_u(p0);    // D-layout -> LDS [i][j]
      ps[wv_][r + hi * 8][16 + lm] = f2bf_u(p1);
    }
#pragma unroll
    for (int t = 0; t < 16; t++)
#pragma unroll
      for (int r = 0; r < 8; r++) O[t][r] *= alpha[r];

    // O += P @ V' (K = 32 j-values), B-frags batched in groups of 4
    v16bf Ap = loadA_bf16(&ps[wv_][lm][0], hi);
#pragma unroll
    for (int tg = 0; tg < 4; tg++) {
      v16bf Bv[4];
#pragma unroll
      for (int t4 = 0; t4 < 4; t4++)
        Bv[t4] = loadB_bf16(&vs[(tg * 4 + t4) * 16 + lm][0], hi);
#pragma unroll
      for (int t4 = 0; t4 < 4; t4++)
        O[tg * 4 + t4] = wmma_bf16(Ap, Bv[t4], O[tg * 4 + t4]);
    }
    __syncthreads();
  }

#pragma unroll
  for (int r = 0; r < 8; r++) l_[r] = 1.f / l_[r];
#pragma unroll
  for (int t = 0; t < 16; t++)
#pragma unroll
    for (int r = 0; r < 8; r++)
      ot[((size_t)b * HW_ + i0 + wv_ * 16 + r + hi * 8) * C_ + t * 16 + lm] =
          f2bf_u(O[t][r] * l_[r]);
}

// y[b][o][p] = sum_c wo[o][c] * ot[b][p][c] + bo[o]   (fp32 out)
__global__ __launch_bounds__(128) void k_out(const u16* __restrict__ ot,
                                             const u16* __restrict__ wb,
                                             const float* __restrict__ bias,
                                             float* __restrict__ y) {
  const int b  = blockIdx.y;
  const int p0 = blockIdx.x * 64;
  const int tid = threadIdx.x;
  const int wv_ = tid >> 5, L = tid & 31, hi = L >> 4, lm = L & 15;

  const u16* orow = ot + ((size_t)b * HW_ + p0 + wv_ * 16 + lm) * C_;
  v16bf Bo[8];
#pragma unroll
  for (int kc = 0; kc < 8; kc++) Bo[kc] = loadB_bf16(orow + kc * 32, hi);

#pragma unroll
  for (int m = 0; m < 16; m++) {
    const u16* wrow = wb + (size_t)(m * 16 + lm) * C_;
    v16bf Aw[8];
#pragma unroll
    for (int kc = 0; kc < 8; kc++) Aw[kc] = loadA_bf16(wrow + kc * 32, hi);
    v8f acc = zero8();
#pragma unroll
    for (int kc = 0; kc < 8; kc++) acc = wmma_bf16(Aw[kc], Bo[kc], acc);
#pragma unroll
    for (int r = 0; r < 8; r++) {
      int o = m * 16 + r + hi * 8;
      y[((size_t)b * C_ + o) * HW_ + p0 + wv_ * 16 + lm] = acc[r] + bias[o];
    }
  }
}

extern "C" void kernel_launch(void* const* d_in, const int* in_sizes, int n_in,
                              void* d_out, int out_size, void* d_ws, size_t ws_size,
                              hipStream_t stream) {
  (void)in_sizes; (void)n_in; (void)out_size; (void)ws_size;
  const float* x  = (const float*)d_in[0];
  const float* Q  = (const float*)d_in[1];
  const float* wq = (const float*)d_in[2];
  const float* bq = (const float*)d_in[3];
  const float* wk = (const float*)d_in[4];
  const float* bk = (const float*)d_in[5];
  const float* wv = (const float*)d_in[6];
  const float* bv = (const float*)d_in[7];
  const float* wo = (const float*)d_in[8];
  const float* bo = (const float*)d_in[9];

  char* ws = (char*)d_ws;
  const size_t SZ = (size_t)B_ * HW_ * C_ * sizeof(u16);   // 8 MB each
  u16* qt  = (u16*)(ws);
  u16* kt  = (u16*)(ws + SZ);
  u16* vv  = (u16*)(ws + 2 * SZ);
  u16* ot  = (u16*)(ws + 3 * SZ);
  u16* wbf = (u16*)(ws + 4 * SZ);                          // 4 x 128 KB bf16 weights
  u16* wqb = wbf;
  u16* wkb = wbf + 65536;
  u16* wvb = wbf + 131072;
  u16* wob = wbf + 196608;

  dim3 g(HW_ / 64, B_), blk(128);
  hipLaunchKernelGGL(k_cvtw, dim3(64), dim3(256), 0, stream, wq, wk, wv, wo, wbf);
  hipLaunchKernelGGL(k_proj, g, blk, 0, stream, Q, wqb, bq, qt, 1, 0.0625f); // q, pre-scaled by 1/sqrt(C)
  hipLaunchKernelGGL(k_proj, g, blk, 0, stream, x, wkb, bk, kt, 1, 1.0f);    // k (transposed layout)
  hipLaunchKernelGGL(k_proj, g, blk, 0, stream, x, wvb, bv, vv, 0, 1.0f);    // v (natural layout)
  hipLaunchKernelGGL(k_attn, g, blk, 0, stream, qt, kt, vv, ot);
  hipLaunchKernelGGL(k_out,  g, blk, 0, stream, ot, wob, bo, (float*)d_out);
}
